// ScOTLayer_17789754540150
// MI455X (gfx1250) — compile-verified, matrix operand order checked by reading, not run on hardware
//
#include <hip/hip_runtime.h>
#include <hip/hip_bf16.h>
#include <cstddef>
#include <cstdint>

#define DI __device__ __forceinline__

typedef __attribute__((ext_vector_type(16))) __bf16        v16bf;
typedef __attribute__((ext_vector_type(8)))  float         v8f;
typedef __attribute__((ext_vector_type(4)))  unsigned int  u32x4;

// Optional gfx1250 async global->LDS path (ASYNCcnt), guarded so absence degrades
// gracefully to the synchronous copy. Probe-confirmed signature: param0 is an
// int4* in addrspace(1) (printed as "__device__"), param1 the LDS-side pointer.
#if defined(__has_builtin)
#if __has_builtin(__builtin_amdgcn_global_load_async_to_lds_b128) && \
    __has_builtin(__builtin_amdgcn_s_wait_asynccnt)
#define USE_ASYNC_LDS 1
#endif
#endif
#ifndef USE_ASYNC_LDS
#define USE_ASYNC_LDS 0
#endif

#if USE_ASYNC_LDS
typedef int v4i_vs __attribute__((vector_size(16)));
typedef __attribute__((address_space(1))) v4i_vs* gv4i_p;   // global int4*
typedef __attribute__((address_space(3))) v4i_vs* lv4i_p;   // LDS int4*
#endif

// ---- problem constants ----
constexpr int Bb   = 8;
constexpr int Hh   = 112;
constexpr int Wimg = 112;
constexpr int Cc   = 192;
constexpr int WS   = 7;
constexpr int SH   = 3;
constexpr int NH   = 6;
constexpr int HD   = 32;
constexpr int NN   = 49;             // tokens per window
constexpr int NWIN = 256;            // 16x16 windows
constexpr int LL   = Bb * Hh * Wimg; // 100352 rows
constexpr int MLP  = 4 * Cc;         // 768
constexpr int CQKV = 3 * Cc;         // 576

// ---- bf16 helpers (round-to-nearest-even) ----
DI unsigned short f2bf(float f) {
    unsigned int u = __float_as_uint(f);
    u += 0x7FFFu + ((u >> 16) & 1u);
    return (unsigned short)(u >> 16);
}
DI float bf2f(unsigned short h) { return __uint_as_float(((unsigned int)h) << 16); }

// ---- WMMA fragment: 16 bf16 per lane, loaded as two 16B chunks from LDS ----
// A (16x32 MxK): lane l: m=l&15, half=l>>4, element e: k=(e>>3)*16 + half*8 + (e&7)
// B (32x16 KxN): lane l: n=l&15, half=l>>4, same k pattern (B staged transposed: row=n, col=k)
union Frag {
    v16bf b;
    u32x4 q[2];
};
DI Frag ldsFrag(const unsigned short* base, int rowBase, int rowStride, int kBase, int lane) {
    const int idx = lane & 15, half = lane >> 4;
    const unsigned short* p = base + (size_t)(rowBase + idx) * rowStride + kBase + half * 8;
    Frag f;
    f.q[0] = *(const u32x4*)(p);
    f.q[1] = *(const u32x4*)(p + 16);
    return f;
}
DI v8f wmma32(const Frag& a, const Frag& b, v8f c) {
    return __builtin_amdgcn_wmma_f32_16x16x32_bf16(false, a.b, false, b.b, (short)0, c, false, false);
}

DI float gelu_tanh(float x) {
    float x3 = x * x * x;
    return 0.5f * x * (1.f + tanhf(0.7978845608028654f * (x + 0.044715f * x3)));
}
// window-order row -> x-order row (window reverse + roll(+3,+3))
DI int win2x(int r) {
    int w = r / NN, i = r - w * NN;
    int b = w >> 8, wi = w & 255, wy = wi >> 4, wx = wi & 15;
    int sy = wy * WS + i / WS, sx = wx * WS + i % WS;
    int y = sy + SH; if (y >= Hh)   y -= Hh;
    int x = sx + SH; if (x >= Wimg) x -= Wimg;
    return b * (Hh * Wimg) + y * Wimg + x;
}

// ---------------- kernel: pack weights to bf16, assemble qkv bias ----------------
__global__ __launch_bounds__(256) void k_pack(
    const float* __restrict__ qw, const float* __restrict__ kw, const float* __restrict__ vw,
    const float* __restrict__ pw, const float* __restrict__ f1w, const float* __restrict__ f2w,
    const float* __restrict__ qb, const float* __restrict__ vb,
    unsigned short* __restrict__ Wqkv, unsigned short* __restrict__ Wproj,
    unsigned short* __restrict__ Wfc1, unsigned short* __restrict__ Wfc2,
    float* __restrict__ bqkv) {
    int idx = blockIdx.x * 256 + threadIdx.x;
    const int n0 = Cc * CQKV;
    const int n1 = n0 + Cc * Cc;
    const int n2 = n1 + Cc * MLP;
    const int n3 = n2 + MLP * Cc;
    const int n4 = n3 + CQKV;
    if (idx < n0) {
        int k = idx / CQKV, n = idx - k * CQKV;
        float v = (n < Cc) ? qw[k * Cc + n]
                           : ((n < 2 * Cc) ? kw[k * Cc + n - Cc] : vw[k * Cc + n - 2 * Cc]);
        Wqkv[idx] = f2bf(v);
    } else if (idx < n1) {
        Wproj[idx - n0] = f2bf(pw[idx - n0]);
    } else if (idx < n2) {
        Wfc1[idx - n1] = f2bf(f1w[idx - n1]);
    } else if (idx < n3) {
        Wfc2[idx - n2] = f2bf(f2w[idx - n2]);
    } else if (idx < n4) {
        int n = idx - n3;
        bqkv[n] = (n < Cc) ? qb[n] : ((n < 2 * Cc) ? 0.f : vb[n - 2 * Cc]);
    }
}

// ---------------- CPB relative-position-bias MLP ----------------
DI float cpb_coord(int d) {            // d in [-6,6]
    float t = (float)d * (8.0f / 6.0f);
    float s = (t > 0.f) ? 1.f : ((t < 0.f) ? -1.f : 0.f);
    return s * log2f(fabsf(t) + 1.f) * (1.0f / 3.0f); // /log2(8)
}
__global__ __launch_bounds__(512) void k_cpb(const float* __restrict__ w0, const float* __restrict__ b0,
                                             const float* __restrict__ w1, float* __restrict__ tab) {
    __shared__ float red[512];
    const int row = blockIdx.x;     // 0..168
    const int j = threadIdx.x;      // 0..511
    float t0 = cpb_coord(row / 13 - 6);
    float t1 = cpb_coord(row % 13 - 6);
    float hid = fmaxf(t0 * w0[j] + t1 * w0[512 + j] + b0[j], 0.f);
    for (int h = 0; h < NH; ++h) {
        red[j] = hid * w1[j * NH + h];
        __syncthreads();
        for (int st = 256; st > 0; st >>= 1) {
            if (j < st) red[j] += red[j + st];
            __syncthreads();
        }
        if (j == 0) tab[row * NH + h] = red[0];
        __syncthreads();
    }
}
// expand to padded (NH, 64, 64): pad cols = -1e30 (softmax kill), pad rows = 0
__global__ __launch_bounds__(256) void k_rpb(const float* __restrict__ tab, float* __restrict__ rpbPad) {
    int idx = blockIdx.x * 256 + threadIdx.x;
    if (idx >= NH * 64 * 64) return;
    int h = idx >> 12, rem = idx & 4095;
    int i = rem >> 6, j = rem & 63;
    float v;
    if (j >= NN) {
        v = -1e30f;
    } else if (i >= NN) {
        v = 0.f;
    } else {
        int dy = i / WS - j / WS + (WS - 1);
        int dx = i % WS - j % WS + (WS - 1);
        float t = tab[(dy * 13 + dx) * NH + h];
        v = 16.f / (1.f + expf(-t));
    }
    rpbPad[idx] = v;
}

// ---------------- roll + window partition + f32->bf16 ----------------
__global__ __launch_bounds__(256) void k_prep(const float* __restrict__ x, unsigned short* __restrict__ xb) {
    int idx = blockIdx.x * 256 + threadIdx.x;
    if (idx >= LL * Cc) return;
    int r = idx / Cc, c = idx - r * Cc;
    int w = r / NN, i = r - w * NN;
    int b = w >> 8, wi = w & 255, wy = wi >> 4, wx = wi & 15;
    int sy = wy * WS + i / WS, sx = wx * WS + i % WS;
    int y = sy + SH; if (y >= Hh)   y -= Hh;     // roll(-SHIFT): h[sy] = x[(sy+3)%H]
    int xx = sx + SH; if (xx >= Wimg) xx -= Wimg;
    size_t src = ((size_t)b * (Hh * Wimg) + (size_t)y * Wimg + xx) * Cc + c;
    xb[idx] = f2bf(x[src]);
}

// ---------------- WMMA GEMM: C[M,N] = A[M,K](bf16) @ Bw[K,N](bf16) + bias ----------------
// 128x64 block tile, 8 waves: wave wv owns row strip wv (16 rows) x all four 16-col tiles.
// A tile staged via async global->LDS (ASYNCcnt) when available; B staged with an
// in-flight transpose (synchronous) so fragment reads stay two contiguous 16B chunks.
// MODE 0: bf16 row-major out   1: f32 out with window->x permutation
// MODE 2: gelu -> bf16 out     3: f32 row-major out
template <int MODE>
__global__ __launch_bounds__(256) void k_gemm(const unsigned short* __restrict__ A,
                                              const unsigned short* __restrict__ Bw,
                                              const float* __restrict__ bias,
                                              void* __restrict__ outp,
                                              int N, int K) {
    __shared__ __align__(16) unsigned short As[128 * 32];  // 128 rows x 32 k (8 KB)
    __shared__ __align__(16) unsigned short Bts[64 * 32];  // transposed: [n][k] (4 KB)
    const int t = threadIdx.x;
    const int lane = t & 31, wv = t >> 5;      // 8 waves
    const int row0 = blockIdx.x * 128, n0 = blockIdx.y * 64;
    v8f acc[4];
#pragma unroll
    for (int j = 0; j < 4; ++j) acc[j] = (v8f){0, 0, 0, 0, 0, 0, 0, 0};

    // A: 128 rows x 4 chunks = 512 chunks -> 2 per thread; B: 32 rows x 8 chunks
    const int ar0 = t >> 2, aseg = t & 3;      // chunks t and t+256
    const int br = t >> 3, bseg = t & 7;
    for (int kk = 0; kk < K; kk += 32) {
        const unsigned short* ap0 = A + (size_t)(row0 + ar0) * K + kk + aseg * 8;
        const unsigned short* ap1 = A + (size_t)(row0 + 64 + ar0) * K + kk + aseg * 8;
        unsigned short* ad0 = As + ar0 * 32 + aseg * 8;
        unsigned short* ad1 = As + (64 + ar0) * 32 + aseg * 8;
#if USE_ASYNC_LDS
        __builtin_amdgcn_global_load_async_to_lds_b128((gv4i_p)ap0, (lv4i_p)ad0, 0, 0);
        __builtin_amdgcn_global_load_async_to_lds_b128((gv4i_p)ap1, (lv4i_p)ad1, 0, 0);
#else
        *(u32x4*)ad0 = *(const u32x4*)ap0;
        *(u32x4*)ad1 = *(const u32x4*)ap1;
#endif
        const unsigned short* bp = Bw + (size_t)(kk + br) * N + n0 + bseg * 8;
        union { u32x4 q; unsigned short s[8]; } bv;
        bv.q = *(const u32x4*)bp;
#pragma unroll
        for (int i = 0; i < 8; ++i) Bts[(bseg * 8 + i) * 32 + br] = bv.s[i];
        if (kk + 32 < K) {
            __builtin_prefetch(ap0 + 32, 0, 1);
            __builtin_prefetch(ap1 + 32, 0, 1);
            __builtin_prefetch(bp + (size_t)32 * N, 0, 1);
        }
#if USE_ASYNC_LDS
        __builtin_amdgcn_s_wait_asynccnt(0);   // own wave's async copies landed in LDS
#endif
        __syncthreads();                        // all waves' tiles visible
        Frag af = ldsFrag(As, wv * 16, 32, 0, lane);
#pragma unroll
        for (int nj = 0; nj < 4; ++nj) {
            Frag bf = ldsFrag(Bts, nj * 16, 32, 0, lane);
            acc[nj] = wmma32(af, bf, acc[nj]);
        }
        __syncthreads();
    }
#pragma unroll
    for (int nj = 0; nj < 4; ++nj) {
        int gn = n0 + nj * 16 + (lane & 15);
        float bs = bias[gn];
#pragma unroll
        for (int r = 0; r < 8; ++r) {
            int gm = row0 + wv * 16 + (lane >> 4) * 8 + r;
            float v = acc[nj][r] + bs;
            if (MODE == 0) {
                ((unsigned short*)outp)[(size_t)gm * N + gn] = f2bf(v);
            } else if (MODE == 1) {
                ((float*)outp)[(size_t)win2x(gm) * N + gn] = v;
            } else if (MODE == 2) {
                ((unsigned short*)outp)[(size_t)gm * N + gn] = f2bf(gelu_tanh(v));
            } else {
                ((float*)outp)[(size_t)gm * N + gn] = v;
            }
        }
    }
}

// region label for shift mask (coordinates in the label image)
DI int regionLab(int sy, int sx) {
    int ry = (sy < Hh - WS) ? 0 : ((sy < Hh - SH) ? 1 : 2);
    int rx = (sx < Wimg - WS) ? 0 : ((sx < Wimg - SH) ? 1 : 2);
    return ry * 3 + rx;
}

// ---------------- attention: one block per (window, head), 4 waves ----------------
__global__ __launch_bounds__(128) void k_attn(const unsigned short* __restrict__ qkv,
                                              const float* __restrict__ rpbPad,
                                              const float* __restrict__ logit_scale,
                                              unsigned short* __restrict__ outb) {
    __shared__ __align__(16) unsigned short qs[64 * 32];   // q rows (normed*scale), padded
    __shared__ __align__(16) unsigned short ks[64 * 32];   // k rows (normed), padded
    __shared__ __align__(16) unsigned short vts[32 * 64];  // v transposed [d][j]
    __shared__ float Ss[64 * 64];                          // bias then logits (f32)
    __shared__ __align__(16) unsigned short Ps[64 * 64];   // softmax probs bf16

    const int t = threadIdx.x;
    const int w = blockIdx.x, h = blockIdx.y;
    const int wi = w & 255, wy = wi >> 4, wx = wi & 15;
    const bool boundary = (wy == 15) || (wx == 15);

    const float scale = expf(fminf(logit_scale[h], 4.6051702f)); // ln(100)

    { // q (threads 0..63) / k (64..127): load 32 bf16, L2-normalize, fold scale into q
        const int isK = t >> 6;
        const int row = t & 63;
        unsigned short* dst = isK ? ks : qs;
        if (row < NN) {
            size_t off = (size_t)(w * NN + row) * CQKV + (isK ? Cc : 0) + h * HD;
            union { u32x4 q[4]; unsigned short s[32]; } rv;
            const u32x4* src = (const u32x4*)(qkv + off);
            rv.q[0] = src[0]; rv.q[1] = src[1]; rv.q[2] = src[2]; rv.q[3] = src[3];
            float f[32]; float ss = 0.f;
#pragma unroll
            for (int d = 0; d < 32; ++d) { f[d] = bf2f(rv.s[d]); ss += f[d] * f[d]; }
            float mul = (isK ? 1.f : scale) / fmaxf(sqrtf(ss), 1e-6f);
#pragma unroll
            for (int d = 0; d < 32; ++d) rv.s[d] = f2bf(f[d] * mul);
            u32x4* dq = (u32x4*)(dst + row * 32);
            dq[0] = rv.q[0]; dq[1] = rv.q[1]; dq[2] = rv.q[2]; dq[3] = rv.q[3];
        } else {
            u32x4 z = {0, 0, 0, 0};
            u32x4* dq = (u32x4*)(dst + row * 32);
            dq[0] = z; dq[1] = z; dq[2] = z; dq[3] = z;
        }
    }
    if (t < 64) { // v: load row, scatter transposed into vts[d][row]
        int row = t;
        if (row < NN) {
            size_t off = (size_t)(w * NN + row) * CQKV + 2 * Cc + h * HD;
            union { u32x4 q[4]; unsigned short s[32]; } rv;
            const u32x4* src = (const u32x4*)(qkv + off);
            rv.q[0] = src[0]; rv.q[1] = src[1]; rv.q[2] = src[2]; rv.q[3] = src[3];
#pragma unroll
            for (int d = 0; d < 32; ++d) vts[d * 64 + row] = rv.s[d];
        } else {
#pragma unroll
            for (int d = 0; d < 32; ++d) vts[d * 64 + row] = 0;
        }
    }
    { // stage bias (padded CPB table + shift mask) into Ss, coalesced
        const float* rp = rpbPad + ((size_t)h << 12);
#pragma unroll
        for (int c = 0; c < 32; ++c) {
            int idx = t + c * 128;
            float v = rp[idx];
            if (boundary) {
                int m = idx >> 6, n = idx & 63;
                if (m < NN && n < NN) {
                    int lm = regionLab(wy * WS + m / WS, wx * WS + m % WS);
                    int ln_ = regionLab(wy * WS + n / WS, wx * WS + n % WS);
                    if (lm != ln_) v -= 100.f;
                }
            }
            Ss[idx] = v;
        }
    }
    __syncthreads();

    const int lane = t & 31, wv = t >> 5;   // 4 waves, wave wv owns row strip wv
    { // S = (q*scale/||q||) @ (k/||k||)^T  + staged bias
        Frag af = ldsFrag(qs, wv * 16, 32, 0, lane);
#pragma unroll
        for (int nj = 0; nj < 4; ++nj) {
            Frag bf = ldsFrag(ks, nj * 16, 32, 0, lane);
            v8f c = {0, 0, 0, 0, 0, 0, 0, 0};
            c = wmma32(af, bf, c);
            const int n = nj * 16 + (lane & 15);
#pragma unroll
            for (int r = 0; r < 8; ++r) {
                int m = wv * 16 + (lane >> 4) * 8 + r;
                Ss[m * 64 + n] += c[r];
            }
        }
    }
    __syncthreads();
    if (t < 64) { // softmax per row -> Ps (bf16); pad cols are -1e30 -> exp 0
        float mx = -1e30f;
        for (int jj = 0; jj < 64; ++jj) mx = fmaxf(mx, Ss[t * 64 + jj]);
        float sum = 0.f;
        for (int jj = 0; jj < 64; ++jj) sum += __expf(Ss[t * 64 + jj] - mx);
        float inv = 1.f / sum;
        for (int jj = 0; jj < 64; ++jj) Ps[t * 64 + jj] = f2bf(__expf(Ss[t * 64 + jj] - mx) * inv);
    }
    __syncthreads();
    { // O = P @ V  (K=64 in two 32-chunks; 2 n-tiles cover HD=32)
        v8f o0 = {0, 0, 0, 0, 0, 0, 0, 0};
        v8f o1 = o0;
#pragma unroll
        for (int kc = 0; kc < 2; ++kc) {
            Frag af = ldsFrag(Ps, wv * 16, 64, kc * 32, lane);
            Frag b0 = ldsFrag(vts, 0, 64, kc * 32, lane);
            Frag b1 = ldsFrag(vts, 16, 64, kc * 32, lane);
            o0 = wmma32(af, b0, o0);
            o1 = wmma32(af, b1, o1);
        }
#pragma unroll
        for (int j = 0; j < 2; ++j) {
            v8f o = j ? o1 : o0;
            int ch = h * HD + j * 16 + (lane & 15);
#pragma unroll
            for (int r = 0; r < 8; ++r) {
                int m = wv * 16 + (lane >> 4) * 8 + r;
                if (m < NN) outb[(size_t)(w * NN + m) * Cc + ch] = f2bf(o[r]);
            }
        }
    }
}

// ---------------- LayerNorm + residual (optionally also emit bf16) ----------------
template <int WRITE_BF>
__global__ __launch_bounds__(192) void k_ln(const float* __restrict__ res,
                                            const float* __restrict__ v,
                                            const float* __restrict__ g,
                                            const float* __restrict__ bb,
                                            float* __restrict__ outF,
                                            unsigned short* __restrict__ outB) {
    __shared__ float s1[256], s2[256];
    const int t = threadIdx.x;
    const size_t r = blockIdx.x;
    float x = v[r * Cc + t];
    s1[t] = x; s2[t] = x * x;
    if (t < 64) { s1[192 + t] = 0.f; s2[192 + t] = 0.f; }
    __syncthreads();
    for (int st = 128; st > 0; st >>= 1) {
        if (t < st) { s1[t] += s1[t + st]; s2[t] += s2[t + st]; }
        __syncthreads();
    }
    float mean = s1[0] * (1.f / 192.f);
    float var = s2[0] * (1.f / 192.f) - mean * mean;
    float y = (x - mean) * rsqrtf(var + 1e-5f) * g[t] + bb[t];
    float o = res[r * Cc + t] + y;
    outF[r * Cc + t] = o;
    if (WRITE_BF) outB[r * Cc + t] = f2bf(o);
}

// ---------------- launch ----------------
extern "C" void kernel_launch(void* const* d_in, const int* in_sizes, int n_in,
                              void* d_out, int out_size, void* d_ws, size_t ws_size,
                              hipStream_t stream) {
    (void)in_sizes; (void)n_in; (void)out_size; (void)ws_size;
    const float* x       = (const float*)d_in[0];
    const float* q_w     = (const float*)d_in[1];
    const float* q_b     = (const float*)d_in[2];
    const float* k_w     = (const float*)d_in[3];
    const float* v_w     = (const float*)d_in[4];
    const float* v_b     = (const float*)d_in[5];
    const float* proj_w  = (const float*)d_in[6];
    const float* proj_b  = (const float*)d_in[7];
    const float* logit_s = (const float*)d_in[8];
    const float* cpb_w0  = (const float*)d_in[9];
    const float* cpb_b0  = (const float*)d_in[10];
    const float* cpb_w1  = (const float*)d_in[11];
    const float* ln1_g   = (const float*)d_in[12];
    const float* ln1_b   = (const float*)d_in[13];
    const float* ln2_g   = (const float*)d_in[14];
    const float* ln2_b   = (const float*)d_in[15];
    const float* fc1_w   = (const float*)d_in[16];
    const float* fc1_b   = (const float*)d_in[17];
    const float* fc2_w   = (const float*)d_in[18];
    const float* fc2_b   = (const float*)d_in[19];
    float* out = (float*)d_out;

    char* ws = (char*)d_ws;
    size_t off = 0;
    auto alloc = [&](size_t bytes) -> void* {
        void* p = ws + off;
        off = (off + bytes + 255) & ~(size_t)255;
        return p;
    };
    unsigned short* xb     = (unsigned short*)alloc((size_t)LL * Cc * 2);   // later reused as attn_out
    unsigned short* qkvb   = (unsigned short*)alloc((size_t)LL * MLP * 2);  // qkv (L,576) then hid (L,768)
    float*          f32b   = (float*)alloc((size_t)LL * Cc * 4);            // proj_out then mlp_out
    float*          h1f    = (float*)alloc((size_t)LL * Cc * 4);
    unsigned short* h1b    = (unsigned short*)alloc((size_t)LL * Cc * 2);
    unsigned short* Wqkv   = (unsigned short*)alloc((size_t)Cc * CQKV * 2);
    unsigned short* Wproj  = (unsigned short*)alloc((size_t)Cc * Cc * 2);
    unsigned short* Wfc1   = (unsigned short*)alloc((size_t)Cc * MLP * 2);
    unsigned short* Wfc2   = (unsigned short*)alloc((size_t)MLP * Cc * 2);
    float*          bqkv   = (float*)alloc((size_t)CQKV * 4);
    float*          tab    = (float*)alloc((size_t)169 * NH * 4);
    float*          rpbPad = (float*)alloc((size_t)NH * 64 * 64 * 4);

    {
        int tot = Cc * CQKV + Cc * Cc + Cc * MLP + MLP * Cc + CQKV;
        k_pack<<<(tot + 255) / 256, 256, 0, stream>>>(q_w, k_w, v_w, proj_w, fc1_w, fc2_w,
                                                      q_b, v_b, Wqkv, Wproj, Wfc1, Wfc2, bqkv);
    }
    k_cpb<<<169, 512, 0, stream>>>(cpb_w0, cpb_b0, cpb_w1, tab);
    {
        int tot = NH * 64 * 64;
        k_rpb<<<(tot + 255) / 256, 256, 0, stream>>>(tab, rpbPad);
    }
    k_prep<<<(LL * Cc + 255) / 256, 256, 0, stream>>>(x, xb);

    // QKV projection: (L,192) @ (192,576) -> bf16 (L,576)
    k_gemm<0><<<dim3(LL / 128, CQKV / 64), 256, 0, stream>>>(xb, Wqkv, bqkv, qkvb, CQKV, Cc);
    // windowed cosine attention -> attn_out (window order, bf16) reusing xb
    k_attn<<<dim3(Bb * NWIN, NH), 128, 0, stream>>>(qkvb, rpbPad, logit_s, xb);
    // output projection, epilogue permutes to x-order f32
    k_gemm<1><<<dim3(LL / 128, Cc / 64), 256, 0, stream>>>(xb, Wproj, proj_b, f32b, Cc, Cc);
    // h1 = x + LN(proj_out)
    k_ln<1><<<LL, 192, 0, stream>>>(x, f32b, ln1_g, ln1_b, h1f, h1b);
    // fc1 + gelu -> bf16 hidden (reuses qkvb region)
    k_gemm<2><<<dim3(LL / 128, MLP / 64), 256, 0, stream>>>(h1b, Wfc1, fc1_b, qkvb, MLP, Cc);
    // fc2 -> f32 (reuses f32b)
    k_gemm<3><<<dim3(LL / 128, Cc / 64), 256, 0, stream>>>(qkvb, Wfc2, fc2_b, f32b, Cc, MLP);
    // out = h1 + LN(mlp)
    k_ln<0><<<LL, 192, 0, stream>>>(h1f, f32b, ln2_g, ln2_b, out, nullptr);
}
